// BiLSTM_CRF_42219528519871
// MI455X (gfx1250) — matrix operand order repaired and checked by reading, not compiled, not run on hardware
//
#include <hip/hip_runtime.h>
#include <hip/hip_bf16.h>

// ---------------------------------------------------------------------------
// Problem constants (from the reference)
// ---------------------------------------------------------------------------
#define S_LEN   1024
#define B_SZ    128
#define E_DIM   256
#define H_DIM   128
#define T_TAGS  7
#define START_TAG 5
#define STOP_TAG  6
#define NEGV    (-10000.0f)

typedef __attribute__((ext_vector_type(16))) __bf16 bf16x16;
typedef __attribute__((ext_vector_type(8)))  float  v8f;

union BF16x16U { bf16x16 v; int4 q[2]; };
union BF16x8U  { int4 q; __bf16 h[8]; };

// ---------------------------------------------------------------------------
// WMMA helpers (CDNA5 16x16x32 bf16 -> f32)
// ---------------------------------------------------------------------------
__device__ __forceinline__ v8f wmma_bf16(bf16x16 a, bf16x16 b, v8f c) {
  // 8 args: (neg_a, A, neg_b, B, c_mod, C, reuse_a, reuse_b)
  return __builtin_amdgcn_wmma_f32_16x16x32_bf16(false, a, false, b,
                                                 (short)0, c, false, false);
}

// A-matrix 16x32 bf16 fragment from row-major (ld = row stride in elems).
// Lane l (l<16): row=l, K chunks {k0..k0+7, k0+16..k0+23}; lanes 16-31: +8.
__device__ __forceinline__ bf16x16 load_frag_a(const __bf16* base, int ld,
                                               int row0, int k0) {
  const int lane = threadIdx.x & 31;
  const __bf16* p = base + (size_t)(row0 + (lane & 15)) * ld
                         + k0 + ((lane >> 4) << 3);
  BF16x16U u;
  u.q[0] = *(const int4*)(p);
  u.q[1] = *(const int4*)(p + 16);
  return u.v;
}

// B-matrix 32x16 fragment: column n = W row n (W row-major NxK, so the
// "NT" GEMM operand is K-contiguous). Lane l: n = n0+(l&15);
// lanes 0-15 hold K chunks {k0..k0+7, k0+16..k0+23}; lanes 16-31: +8.
__device__ __forceinline__ bf16x16 load_frag_b(const __bf16* W, int ld,
                                               int n0, int k0) {
  const int lane = threadIdx.x & 31;
  const __bf16* p = W + (size_t)(n0 + (lane & 15)) * ld
                      + k0 + ((lane >> 4) << 3);
  BF16x16U u;
  u.q[0] = *(const int4*)(p);
  u.q[1] = *(const int4*)(p + 16);
  return u.v;
}

__device__ __forceinline__ float sigm(float x) {
  return 1.0f / (1.0f + __expf(-x));
}

// ---------------------------------------------------------------------------
// Utility kernels
// ---------------------------------------------------------------------------
__global__ void f2bf_kernel(const float* __restrict__ in,
                            __bf16* __restrict__ out, int n) {
  int i = blockIdx.x * blockDim.x + threadIdx.x;
  if (i < n) out[i] = (__bf16)in[i];
}

__global__ void bias_sum_kernel(const float* __restrict__ bih,
                                const float* __restrict__ bhh,
                                float* __restrict__ bsum, int n) {
  int i = blockIdx.x * blockDim.x + threadIdx.x;
  if (i < n) bsum[i] = bih[i] + bhh[i];
}

// Embedding gather + fp32->bf16. One thread = 4 elements of one row.
__global__ void embed_kernel(const int* __restrict__ sent,
                             const float* __restrict__ table,
                             __bf16* __restrict__ X) {
  size_t idx = (size_t)blockIdx.x * blockDim.x + threadIdx.x;
  size_t row = idx >> 6;                 // (s*B + b)
  int e0 = (int)(idx & 63) << 2;
  int tok = sent[row];
  float4 f = *(const float4*)(table + (size_t)tok * E_DIM + e0);
  __bf16* xp = X + row * E_DIM + e0;
  xp[0] = (__bf16)f.x; xp[1] = (__bf16)f.y;
  xp[2] = (__bf16)f.z; xp[3] = (__bf16)f.w;
}

// ---------------------------------------------------------------------------
// Pregate GEMM: G(M,1024) = X(M,256) @ W(1024,256)^T + bias
// Block = 256 threads (8 waves). Block tile 64(M) x 256(N); each wave keeps
// 8 accumulator tiles (16x128 strip). Per k-step: one clause of 9 fragment
// loads (1 A + 8 B), then 8 back-to-back WMMAs -> staggered s_wait_loadcnt
// instead of wait-to-zero per WMMA.
// ---------------------------------------------------------------------------
__global__ __launch_bounds__(256) void pregate_gemm_kernel(
    const __bf16* __restrict__ X,    // (M, 256)
    const __bf16* __restrict__ W,    // (1024, 256) = w_ih[l] both dirs
    const float*  __restrict__ bias, // (1024)
    float*        __restrict__ G) {  // (M, 1024)
  const int wave = threadIdx.x >> 5;
  const int lane = threadIdx.x & 31;
  const int mt = wave & 3;           // 4 M-tiles
  const int ng = wave >> 2;          // 2 N-groups of 128
  const int row0 = blockIdx.x * 64 + mt * 16;
  const int col0 = blockIdx.y * 256 + ng * 128;

  v8f acc[8];
#pragma unroll
  for (int j = 0; j < 8; ++j)
#pragma unroll
    for (int v = 0; v < 8; ++v) acc[j][v] = 0.0f;

#pragma unroll
  for (int k0 = 0; k0 < E_DIM; k0 += 32) {
    // Phase 1: issue the whole fragment clause for this k-step.
    bf16x16 a = load_frag_a(X, E_DIM, row0, k0);
    bf16x16 bfrag[8];
#pragma unroll
    for (int j = 0; j < 8; ++j)
      bfrag[j] = load_frag_b(W, E_DIM, col0 + j * 16, k0);
    // Phase 2: 8 back-to-back WMMAs on the loaded fragments.
#pragma unroll
    for (int j = 0; j < 8; ++j)
      acc[j] = wmma_bf16(a, bfrag[j], acc[j]);
  }

  const int nrel = lane & 15;
  const int mrel = (lane >> 4) << 3;
#pragma unroll
  for (int j = 0; j < 8; ++j) {
    int col = col0 + j * 16 + nrel;
    float bs = bias[col];
#pragma unroll
    for (int v = 0; v < 8; ++v)
      G[(size_t)(row0 + mrel + v) * 1024 + col] = acc[j][v] + bs;
  }
}

// ---------------------------------------------------------------------------
// Persistent LSTM recurrence. gridDim.x = 2 (dir), 1024 threads = 32 waves.
// h_{t-1} in LDS (bf16, padded stride 136 elems = 272B to spread banks);
// c stays in registers: each wave owns (m-tile, 2 column-units) and computes
// all four gates for those (b,h) positions, so i/f/g/o line up per-lane.
// Pregates for t+1 are prefetched (global_prefetch_b8) during step t.
// ---------------------------------------------------------------------------
__global__ __launch_bounds__(1024) void lstm_rec_kernel(
    const float*  __restrict__ G,    // (S*B, 1024) pregates (both dirs)
    const __bf16* __restrict__ Whh,  // (L,2,512,128) bf16
    const float*  __restrict__ h0,   // (2L, B, H)
    const float*  __restrict__ c0,   // (2L, B, H)
    __bf16*       __restrict__ Xout, // (S*B, 256) concat [hf, hb]
    int layer) {
  const int dir = blockIdx.x;
  __shared__ __align__(16) __bf16 hbuf[128][136];

  const int tid  = threadIdx.x;
  const int wave = tid >> 5, lane = tid & 31;
  const int mt = wave & 7;           // 8 M-tiles over B=128
  const int cg = wave >> 3;          // 4 column-groups of 32 over H=128
  const int row0 = mt * 16;
  const int nrel = lane & 15;
  const int mrel = (lane >> 4) << 3;

  const __bf16* Wd  = Whh + (size_t)(layer * 2 + dir) * 512 * 128;
  const float*  h0d = h0  + (size_t)(2 * layer + dir) * B_SZ * H_DIM;
  const float*  c0d = c0  + (size_t)(2 * layer + dir) * B_SZ * H_DIM;

  for (int i = tid; i < B_SZ * H_DIM; i += 1024)
    hbuf[i >> 7][i & 127] = (__bf16)h0d[i];

  float cst[2][8];
#pragma unroll
  for (int u = 0; u < 2; ++u) {
    int col = cg * 32 + u * 16 + nrel;
#pragma unroll
    for (int v = 0; v < 8; ++v)
      cst[u][v] = c0d[(row0 + mrel + v) * H_DIM + col];
  }
  __syncthreads();

  for (int t = 0; t < S_LEN; ++t) {
    const int tt = dir ? (S_LEN - 1 - t) : t;
    const size_t gbase = (size_t)tt * B_SZ;
    float hv[2][8];

    // Prefetch next step's pregate rows into cache while we compute.
    if (t + 1 < S_LEN) {
      const int tn = dir ? (S_LEN - 2 - t) : (t + 1);
      const float* gp = G + ((size_t)tn * B_SZ + row0 + mrel) * 1024
                          + dir * 512 + cg * 32 + nrel;
      __builtin_prefetch(gp, 0, 1);
      __builtin_prefetch(gp + 4 * 1024, 0, 1);
    }

#pragma unroll
    for (int u = 0; u < 2; ++u) {
      const int col0 = cg * 32 + u * 16;
      const int col  = col0 + nrel;
      v8f acc[4];
      // C operand = pregate (x@Wih^T + biases), so D = h@Whh^T + pregate
#pragma unroll
      for (int g = 0; g < 4; ++g)
#pragma unroll
        for (int v = 0; v < 8; ++v)
          acc[g][v] = G[(gbase + row0 + mrel + v) * 1024
                        + dir * 512 + g * 128 + col];

#pragma unroll
      for (int k0 = 0; k0 < H_DIM; k0 += 32) {
        // Clause: A from LDS + all four gate B-fragments, then 4 WMMAs.
        bf16x16 a = load_frag_a(&hbuf[0][0], 136, row0, k0);
        bf16x16 bb[4];
#pragma unroll
        for (int g = 0; g < 4; ++g)
          bb[g] = load_frag_b(Wd, H_DIM, g * 128 + col0, k0);
#pragma unroll
        for (int g = 0; g < 4; ++g)
          acc[g] = wmma_bf16(a, bb[g], acc[g]);
      }
#pragma unroll
      for (int v = 0; v < 8; ++v) {
        float iv = sigm(acc[0][v]);
        float fv = sigm(acc[1][v]);
        float gv = tanhf(acc[2][v]);
        float ov = sigm(acc[3][v]);
        float c  = fv * cst[u][v] + iv * gv;
        cst[u][v] = c;
        hv[u][v] = ov * tanhf(c);
      }
    }

    __syncthreads();  // everyone done reading h_{t-1}
#pragma unroll
    for (int u = 0; u < 2; ++u) {
      int col = cg * 32 + u * 16 + nrel;
#pragma unroll
      for (int v = 0; v < 8; ++v) {
        int b = row0 + mrel + v;
        __bf16 hb = (__bf16)hv[u][v];
        hbuf[b][col] = hb;
        Xout[(gbase + b) * 256 + dir * 128 + col] = hb;
      }
    }
    __syncthreads();  // h_t visible before next step
  }
}

// ---------------------------------------------------------------------------
// Output projection: feats(S*B,7) = X2 @ w_out^T + b_out. One wave per row.
// ---------------------------------------------------------------------------
__global__ __launch_bounds__(256) void feats_kernel(
    const __bf16* __restrict__ X2,   // (S*B, 256)
    const float*  __restrict__ Wout, // (7, 256)
    const float*  __restrict__ bout, // (7)
    float*        __restrict__ feats) {
  const int wave = threadIdx.x >> 5, lane = threadIdx.x & 31;
  const size_t row = (size_t)blockIdx.x * 8 + wave;
  BF16x8U u;
  u.q = *(const int4*)(X2 + row * 256 + lane * 8);
  float xv[8];
#pragma unroll
  for (int j = 0; j < 8; ++j) xv[j] = (float)u.h[j];

#pragma unroll
  for (int tag = 0; tag < T_TAGS; ++tag) {
    const float* wp = Wout + tag * 256 + lane * 8;
    float p = 0.0f;
#pragma unroll
    for (int j = 0; j < 8; ++j) p += xv[j] * wp[j];
    p += __shfl_xor(p, 16, 32);
    p += __shfl_xor(p, 8, 32);
    p += __shfl_xor(p, 4, 32);
    p += __shfl_xor(p, 2, 32);
    p += __shfl_xor(p, 1, 32);
    if (lane == 0) feats[row * T_TAGS + tag] = p + bout[tag];
  }
}

// ---------------------------------------------------------------------------
// Viterbi decode: one thread per batch element (7x7 transition table).
// d_out = [path (B,S) as float][score (B)]
// ---------------------------------------------------------------------------
__global__ void viterbi_kernel(const float* __restrict__ feats,  // (S,B,7)
                               const float* __restrict__ trans,  // (7,7)
                               float* __restrict__ out,
                               unsigned char* __restrict__ bptr) {
  int b = blockIdx.x * blockDim.x + threadIdx.x;
  if (b >= B_SZ) return;

  float tr[T_TAGS * T_TAGS];
#pragma unroll
  for (int i = 0; i < T_TAGS * T_TAGS; ++i) tr[i] = trans[i];

  float fv[T_TAGS];
#pragma unroll
  for (int i = 0; i < T_TAGS; ++i) fv[i] = NEGV;
  fv[START_TAG] = 0.0f;

  for (int s = 0; s < S_LEN; ++s) {
    const float* fp = feats + ((size_t)s * B_SZ + b) * T_TAGS;
    float nf[T_TAGS];
#pragma unroll
    for (int nx = 0; nx < T_TAGS; ++nx) {
      float best = fv[0] + tr[nx * T_TAGS + 0];
      int arg = 0;
#pragma unroll
      for (int pv = 1; pv < T_TAGS; ++pv) {
        float v = fv[pv] + tr[nx * T_TAGS + pv];
        if (v > best) { best = v; arg = pv; }
      }
      nf[nx] = best + fp[nx];
      bptr[((size_t)b * S_LEN + s) * T_TAGS + nx] = (unsigned char)arg;
    }
#pragma unroll
    for (int i = 0; i < T_TAGS; ++i) fv[i] = nf[i];
  }

  float best = fv[0] + tr[STOP_TAG * T_TAGS + 0];
  int arg = 0;
#pragma unroll
  for (int pv = 1; pv < T_TAGS; ++pv) {
    float v = fv[pv] + tr[STOP_TAG * T_TAGS + pv];
    if (v > best) { best = v; arg = pv; }
  }
  out[(size_t)B_SZ * S_LEN + b] = best;

  int tag = arg;
  for (int s = S_LEN - 1; s >= 0; --s) {
    out[(size_t)b * S_LEN + s] = (float)tag;
    tag = bptr[((size_t)b * S_LEN + s) * T_TAGS + tag];
  }
}

// ---------------------------------------------------------------------------
// Host launcher
// ---------------------------------------------------------------------------
extern "C" void kernel_launch(void* const* d_in, const int* in_sizes, int n_in,
                              void* d_out, int out_size, void* d_ws,
                              size_t ws_size, hipStream_t stream) {
  const int*   sentence = (const int*)d_in[0];
  const float* embed_t  = (const float*)d_in[1];
  const float* w_ih     = (const float*)d_in[2];   // (2,2,512,256)
  const float* w_hh     = (const float*)d_in[3];   // (2,2,512,128)
  const float* b_ih     = (const float*)d_in[4];   // (2,2,512)
  const float* b_hh     = (const float*)d_in[5];
  const float* h0       = (const float*)d_in[6];   // (4,128,128)
  const float* c0       = (const float*)d_in[7];
  const float* w_out    = (const float*)d_in[8];   // (7,256)
  const float* b_out    = (const float*)d_in[9];   // (7)
  const float* trans    = (const float*)d_in[10];  // (7,7)
  float* out = (float*)d_out;

  const size_t SB = (size_t)S_LEN * B_SZ;          // 131072
  char* ws = (char*)d_ws;
  size_t off = 0;
  auto alloc = [&](size_t bytes) -> void* {
    void* p = ws + off;
    off += (bytes + 255) & ~(size_t)255;
    return p;
  };
  __bf16* xA   = (__bf16*)alloc(SB * E_DIM * sizeof(__bf16));  // 64 MB
  __bf16* xB   = (__bf16*)alloc(SB * E_DIM * sizeof(__bf16));  // 64 MB
  float*  G    = (float*)alloc(SB * 1024 * sizeof(float));     // 512 MB
  __bf16* wihb = (__bf16*)alloc((size_t)524288 * sizeof(__bf16));
  __bf16* whhb = (__bf16*)alloc((size_t)262144 * sizeof(__bf16));
  float*  bias = (float*)alloc(2048 * sizeof(float));
  float*  feats= (float*)alloc(SB * T_TAGS * sizeof(float));
  unsigned char* bptr = (unsigned char*)alloc(SB * T_TAGS);

  // Weight / bias conversion
  f2bf_kernel<<<524288 / 256, 256, 0, stream>>>(w_ih, wihb, 524288);
  f2bf_kernel<<<262144 / 256, 256, 0, stream>>>(w_hh, whhb, 262144);
  bias_sum_kernel<<<2048 / 256, 256, 0, stream>>>(b_ih, b_hh, bias, 2048);

  // Embedding -> xA (bf16)
  embed_kernel<<<(unsigned)(SB * 64 / 256), 256, 0, stream>>>(sentence,
                                                              embed_t, xA);

  for (int l = 0; l < 2; ++l) {
    const __bf16* xin  = (l == 0) ? xA : xB;
    __bf16*       xout = (l == 0) ? xB : xA;
    // G = xin @ w_ih[l]^T + (b_ih+b_hh)[l], both directions fused (N=1024)
    pregate_gemm_kernel<<<dim3(2048, 4), 256, 0, stream>>>(
        xin, wihb + (size_t)l * 262144, bias + (size_t)l * 1024, G);
    // persistent recurrence, one block per direction
    lstm_rec_kernel<<<2, 1024, 0, stream>>>(G, whhb, h0, c0, xout, l);
  }

  // Final layer output is in xA (layer 1 wrote xA)
  feats_kernel<<<(unsigned)(SB / 8), 256, 0, stream>>>(xA, w_out, b_out,
                                                       feats);
  viterbi_kernel<<<1, 128, 0, stream>>>(feats, trans, out, bptr);
}
